// H2HGCN_56788057588085
// MI455X (gfx1250) — compile-verified, matrix operand order checked by reading, not run on hardware
//
#include <hip/hip_runtime.h>
#include <hip/hip_bf16.h>
#include <math.h>

// ---------------------------------------------------------------------------
// H2HGCN forward for MI455X (gfx1250, wave32, WMMA).
//   N=30000 nodes, K=32 neighbors, F=256 in-feats, D=128 hidden.
// GEMMs via v_wmma_f32_16x16x32_f16 (fp32 accumulate). Weight matrices are
// pre-packed once (transposed + f16) so B fragments load as b128 clauses.
// msg buffer (15.4MB) stays L2-resident (192MB L2) for the neighbor gather.
// ---------------------------------------------------------------------------

#define NN 30000
#define KK 32
#define FF 256
#define DD 128

typedef __attribute__((ext_vector_type(16))) _Float16 v16h;
typedef __attribute__((ext_vector_type(8)))  float    v8f;

#define SELU_SCALE 1.0507009873554805f
#define SELU_ALPHA 1.6732632423543772f

__device__ __forceinline__ float selu_f(float x) {
  return SELU_SCALE * (x > 0.f ? x : SELU_ALPHA * (__expf(x) - 1.f));
}

// A fragment: 16x32 f16, lane (l%16) = row, lane/16 selects K-interleave half.
// v16h elems j=0..7  -> K = kbase + 8*half + j          (contiguous in memory)
//           j=8..15 -> K = kbase + 16 + 8*half + (j-8)  (contiguous in memory)
__device__ __forceinline__ v16h load_a_frag(const float* __restrict__ A, int lda,
                                            int row, int kbase, int half) {
  v16h a;
  const float* p0 = A + (size_t)row * lda + kbase + 8 * half;
  const float* p1 = p0 + 16;
#pragma unroll
  for (int j = 0; j < 8; ++j) {
    a[j]     = (_Float16)p0[j];
    a[8 + j] = (_Float16)p1[j];
  }
  return a;
}

// B fragment from pre-packed transposed f16 weights Bt[col][k] (ldk = K dim).
// Same K interleave as A; two contiguous 16-byte reads per lane.
__device__ __forceinline__ v16h load_bT_frag(const _Float16* __restrict__ Bt,
                                             int ldk, int col, int kbase, int half) {
  v16h b;
  const _Float16* p0 = Bt + (size_t)col * ldk + kbase + 8 * half;
  const _Float16* p1 = p0 + 16;
#pragma unroll
  for (int j = 0; j < 8; ++j) {
    b[j]     = p0[j];
    b[8 + j] = p1[j];
  }
  return b;
}

// ---------------------------------------------------------------------------
// Kernel 1: h = normalize(exp_map_zero(selu(x @ lin_W + b)))
// Block: 256 threads (8 waves). Each block: 16 rows x 128 cols.
// Wave w owns cols [16w, 16w+16). K=256 consumed in 8 WMMA steps.
// ---------------------------------------------------------------------------
__global__ __launch_bounds__(256) void gemm_selu_expmap_kernel(
    const float* __restrict__ x, const _Float16* __restrict__ WT,
    const float* __restrict__ bias, float* __restrict__ h) {
  __shared__ float smem[16 * 132];  // 16 rows x 128 cols, padded

  const int wave = threadIdx.x >> 5;
  const int lane = threadIdx.x & 31;
  const int half = lane >> 4;
  const int lr   = lane & 15;
  const int row0 = blockIdx.x * 16;
  const int col  = wave * 16 + lr;

  v8f acc = {};
#pragma unroll
  for (int kk = 0; kk < FF; kk += 32) {
    v16h a = load_a_frag(x, FF, row0 + lr, kk, half);
    v16h b = load_bT_frag(WT, FF, col, kk, half);
    acc = __builtin_amdgcn_wmma_f32_16x16x32_f16(
        false, a, false, b, (short)0, acc, false, false);
  }

  // Epilogue part 1: bias + selu into LDS tile.
  // C layout: elem r -> row (r + 8*half), col = lane%16 within wave slice.
  const float bv = bias[col];
#pragma unroll
  for (int r = 0; r < 8; ++r) {
    int m = r + 8 * half;
    smem[m * 132 + col] = selu_f(acc[r] + bv);
  }
  __syncthreads();

  // Epilogue part 2: exp_map_zero + normalize_lorentz, 16 threads per row.
  const int row = threadIdx.x >> 4;  // 0..15
  const int seg = threadIdx.x & 15;  // 8 cols per thread
  float part = 0.f;
#pragma unroll
  for (int c = 0; c < 8; ++c) {
    int d = seg * 8 + c;
    float v = smem[row * 132 + d];
    if (d >= 1) part += v * v;  // ldv over space components only
  }
#pragma unroll
  for (int m = 1; m < 16; m <<= 1) part += __shfl_xor(part, m, 32);
  const float ldv = part;

  const float nd = sqrtf(fmaxf(ldv + 1e-9f, 1e-10f));
  const float t  = fminf(nd, 1.0f);  // NORM_CLIP
  const float e  = __expf(t);
  const float sh = 0.5f * (e - 1.f / e);   // sinh(t)
  const float sc = sh / nd;                // space scale
  // normalize_lorentz recomputes time from scaled space: sum(space^2)=sc^2*ldv
  const float time = sqrtf(1.f + sc * sc * ldv);

  float* out = h + (size_t)(row0 + row) * DD;
#pragma unroll
  for (int c = 0; c < 8; ++c) {
    int d = seg * 8 + c;
    out[d] = (d == 0) ? time : sc * smem[row * 132 + d];
  }
}

// ---------------------------------------------------------------------------
// Kernel 2: msg = h @ lw   (128x128, K consumed in 4 WMMA steps)
// ---------------------------------------------------------------------------
__global__ __launch_bounds__(256) void gemm_msg_kernel(
    const float* __restrict__ h, const _Float16* __restrict__ lwT,
    float* __restrict__ msg) {
  const int wave = threadIdx.x >> 5;
  const int lane = threadIdx.x & 31;
  const int half = lane >> 4;
  const int lr   = lane & 15;
  const int row0 = blockIdx.x * 16;
  const int col  = wave * 16 + lr;

  v8f acc = {};
#pragma unroll
  for (int kk = 0; kk < DD; kk += 32) {
    v16h a = load_a_frag(h, DD, row0 + lr, kk, half);
    v16h b = load_bT_frag(lwT, DD, col, kk, half);
    acc = __builtin_amdgcn_wmma_f32_16x16x32_f16(
        false, a, false, b, (short)0, acc, false, false);
  }
#pragma unroll
  for (int r = 0; r < 8; ++r) {
    int m = r + 8 * half;
    msg[(size_t)(row0 + m) * DD + col] = acc[r];
  }
}

// ---------------------------------------------------------------------------
// Kernel 3: hyperbolic weighted mean + lorentz activation + normalize.
// One 128-thread block per node. msg (15.4MB) is L2-resident -> gather fast.
// ---------------------------------------------------------------------------
__device__ __forceinline__ float block_reduce_128(float v, float* sred) {
#pragma unroll
  for (int m = 1; m < 32; m <<= 1) v += __shfl_xor(v, m, 32);
  const int wid = threadIdx.x >> 5;
  __syncthreads();  // guard sred reuse
  if ((threadIdx.x & 31) == 0) sred[wid] = v;
  __syncthreads();
  return sred[0] + sred[1] + sred[2] + sred[3];
}

__global__ __launch_bounds__(128) void agg_kernel(
    const float* __restrict__ msg, const int* __restrict__ adj,
    const float* __restrict__ w, float* __restrict__ hout) {
  __shared__ float sm[KK * 129];  // 32 neighbor rows, padded (conflict-free)
  __shared__ float slam[KK];
  __shared__ float sinv[KK];
  __shared__ float sw[KK];
  __shared__ int   sadj[KK];
  __shared__ float sred[4];

  const int n = blockIdx.x;
  const int t = threadIdx.x;

  // Stage adjacency + weights once (lanes 0..31), then 32 neighbor rows
  // (coalesced 128-float lines, mostly L2 hits).
  if (t < KK) {
    sadj[t] = adj[n * KK + t];
    sw[t]   = w[n * KK + t];
  }
  __syncthreads();
#pragma unroll 4
  for (int j = 0; j < KK; ++j) {
    sm[j * 129 + t] = msg[(size_t)sadj[j] * DD + t];
  }
  __syncthreads();

  // Per-neighbor Klein norm + lambda weight (lanes 0..31).
  if (t < KK) {
    float inv0 = 1.f / sm[t * 129 + 0];
    float s = 0.f;
    for (int d = 1; d < DD; ++d) {
      float k = sm[t * 129 + d] * inv0;
      s += k * k;
    }
    float kn = fminf(fmaxf(s, 0.f), 0.9f);
    slam[t] = sw[t] * rsqrtf(1.f - kn);
    sinv[t] = inv0;
  }
  __syncthreads();

  // Weighted Klein mean, one dimension per thread.
  float lamsum = 0.f;
#pragma unroll
  for (int j = 0; j < KK; ++j) lamsum += slam[j];

  const int d = t;
  float km = 0.f;
  if (d >= 1) {
    float s = 0.f;
#pragma unroll 4
    for (int j = 0; j < KK; ++j) s += slam[j] * sinv[j] * sm[j * 129 + d];
    km = s / lamsum;
  }

  float kn2 = block_reduce_128(km * km, sred);
  kn2 = fminf(kn2, 0.9f);
  const float root = sqrtf(1.f - kn2);

  // Lorentz activation: h = [1,km]/root ; p = h_sp/(h_t+1) = km/(1+root)
  const float p  = (d >= 1) ? selu_f(km / (1.f + root)) : 0.f;
  const float pn = block_reduce_128(p * p, sred);

  const float denom = 1.f - pn + 1e-6f;          // EPS
  const float spd   = 2.f * p / denom;           // space components
  // normalize_lorentz: sum(sp^2) = 4*pn/denom^2 analytically
  const float time  = sqrtf(1.f + 4.f * pn / (denom * denom));

  hout[(size_t)n * DD + d] = (d == 0) ? time : spd;
}

// ---------------------------------------------------------------------------
// Kernel 0a: pack WT[c][k] = f16(lin_W[k][c])   (128 x 256 f16)
// ---------------------------------------------------------------------------
__global__ __launch_bounds__(256) void pack_WT_kernel(
    const float* __restrict__ W, _Float16* __restrict__ WT) {
  int i = blockIdx.x * blockDim.x + threadIdx.x;  // i = c*FF + k
  if (i >= DD * FF) return;
  int c = i >> 8, k = i & (FF - 1);
  WT[i] = (_Float16)W[(size_t)k * DD + c];
}

// ---------------------------------------------------------------------------
// Kernel 0b: lwT[c][k] = f16(blockdiag(1, M)[k][c])   (128 x 128 f16)
// ---------------------------------------------------------------------------
__global__ __launch_bounds__(256) void build_lwT_kernel(
    const float* __restrict__ M, _Float16* __restrict__ lwT) {
  int i = blockIdx.x * blockDim.x + threadIdx.x;  // i = c*DD + k
  if (i >= DD * DD) return;
  int c = i >> 7, k = i & (DD - 1);
  float v = 0.f;
  if (k == 0 && c == 0) v = 1.f;
  else if (k > 0 && c > 0) v = M[(k - 1) * (DD - 1) + (c - 1)];
  lwT[i] = (_Float16)v;
}

// ---------------------------------------------------------------------------
extern "C" void kernel_launch(void* const* d_in, const int* in_sizes, int n_in,
                              void* d_out, int out_size, void* d_ws, size_t ws_size,
                              hipStream_t stream) {
  const float* x     = (const float*)d_in[0];
  const int*   adj   = (const int*)d_in[1];
  const float* w     = (const float*)d_in[2];
  const float* lin_W = (const float*)d_in[3];
  const float* lin_b = (const float*)d_in[4];
  const float* M     = (const float*)d_in[5];

  float*    h   = (float*)d_out;                    // N x 128 (ping buffer = output)
  float*    msg = (float*)d_ws;                     // N x 128 f32
  _Float16* WT  = (_Float16*)(msg + (size_t)NN * DD);   // 128 x 256 f16
  _Float16* lwT = WT + (size_t)DD * FF;                 // 128 x 128 f16

  pack_WT_kernel<<<(DD * FF + 255) / 256, 256, 0, stream>>>(lin_W, WT);
  build_lwT_kernel<<<(DD * DD + 255) / 256, 256, 0, stream>>>(M, lwT);
  gemm_selu_expmap_kernel<<<NN / 16, 256, 0, stream>>>(x, WT, lin_b, h);

  for (int layer = 0; layer < 2; ++layer) {
    gemm_msg_kernel<<<NN / 16, 256, 0, stream>>>(h, lwT, msg);
    agg_kernel<<<NN, 128, 0, stream>>>(msg, adj, w, h);
  }
}
